// SurfaceEmbedding_9156870275551
// MI455X (gfx1250) — matrix operand order repaired and checked by reading, not compiled, not run on hardware
//
#include <hip/hip_runtime.h>
#include <hip/hip_bf16.h>

typedef __attribute__((ext_vector_type(16))) _Float16 v16h;
typedef __attribute__((ext_vector_type(8)))  float    v8f;

#define BB 4
#define SS 512
#define QQ 256
#define NN 768      // SS + QQ
#define DD 64
#define EPSF 1e-5f
#define SQRT2F 1.41421356237309515f
#define LOG2EF 1.44269504088896340736f

// workspace layout (floats):
//   [0,            BB*NN)     norm_lm  (surf rows 0..511, query rows 512..767 per b)
//   [BB*NN,      2*BB*NN)     norm_tm
//   [2*BB*NN,    2*BB*NN+DD)  alphaL[d] = -scale0/(trace*2*bw^2) * log2(e)
//   [...,        ...+DD)      betaL[d]  = -scale1/(trace*2*bw^2) * log2(e)
// total 6272 floats = 25088 bytes

__device__ __forceinline__ float erfinv_f(float x) {
  // Giles (2010) single-precision approximation
  float w = -__logf((1.0f - x) * (1.0f + x));
  float p;
  if (w < 5.0f) {
    w -= 2.5f;
    p = 2.81022636e-08f;
    p = fmaf(p, w, 3.43273939e-07f);
    p = fmaf(p, w, -3.5233877e-06f);
    p = fmaf(p, w, -4.39150654e-06f);
    p = fmaf(p, w, 0.00021858087f);
    p = fmaf(p, w, -0.00125372503f);
    p = fmaf(p, w, -0.00417768164f);
    p = fmaf(p, w, 0.246640727f);
    p = fmaf(p, w, 1.50140941f);
  } else {
    w = sqrtf(w) - 3.0f;
    p = -0.000200214257f;
    p = fmaf(p, w, 0.000100950558f);
    p = fmaf(p, w, 0.00134934322f);
    p = fmaf(p, w, -0.00367342844f);
    p = fmaf(p, w, 0.00573950773f);
    p = fmaf(p, w, -0.0076224613f);
    p = fmaf(p, w, 0.00943887047f);
    p = fmaf(p, w, 1.00167406f);
    p = fmaf(p, w, 2.83297682f);
  }
  return p * x;
}

// ---------------- Stage 1: stats, normalized coords, per-d constants, ext ----
__global__ __launch_bounds__(256)
void se_stage1(const float* __restrict__ surf_lm, const float* __restrict__ surf_ttm,
               const float* __restrict__ q_lm,    const float* __restrict__ q_ttm,
               const float* __restrict__ mkt_ret, const float* __restrict__ mkt_vol,
               const float* __restrict__ tsy,     const float* __restrict__ tvm,
               const float* __restrict__ tvs,     const float* __restrict__ kls,
               float* __restrict__ ws, float* __restrict__ ext_out) {
  __shared__ float r0[256], r1[256], r2[256], r3[256];
  __shared__ float st[4]; // mean_lm, rstd_lm, mean_tm, rstd_tm
  const int tid = threadIdx.x;

  float slm = 0.f, slm2 = 0.f, stm = 0.f, stm2 = 0.f;
  for (int i = tid; i < BB * SS; i += 256) {
    float x = surf_lm[i];  slm += x; slm2 += x * x;
    float y = surf_ttm[i]; stm += y; stm2 += y * y;
  }
  for (int i = tid; i < BB * QQ; i += 256) {
    float x = q_lm[i];  slm += x; slm2 += x * x;
    float y = q_ttm[i]; stm += y; stm2 += y * y;
  }
  r0[tid] = slm; r1[tid] = slm2; r2[tid] = stm; r3[tid] = stm2;
  __syncthreads();
  for (int off = 128; off >= 1; off >>= 1) {
    if (tid < off) {
      r0[tid] += r0[tid + off]; r1[tid] += r1[tid + off];
      r2[tid] += r2[tid + off]; r3[tid] += r3[tid + off];
    }
    __syncthreads();
  }
  if (tid == 0) {
    const float inv_n = 1.0f / (float)(BB * NN);
    float m = r0[0] * inv_n;               st[0] = m;
    float v = r1[0] * inv_n - m * m;       st[1] = rsqrtf(v + EPSF);
    m = r2[0] * inv_n;                     st[2] = m;
    v = r3[0] * inv_n - m * m;             st[3] = rsqrtf(v + EPSF);
  }
  __syncthreads();
  const float mlm = st[0], rlm = st[1], mtm = st[2], rtm = st[3];

  // normalized coords for all (b, n): surf first, then queries
  for (int i = tid; i < BB * NN; i += 256) {
    int b = i / NN, n = i % NN;
    float lm = (n < SS) ? surf_lm[b * SS + n] : q_lm[b * QQ + (n - SS)];
    float tm = (n < SS) ? surf_ttm[b * SS + n] : q_ttm[b * QQ + (n - SS)];
    ws[i]           = (lm - mlm) * rlm;
    ws[BB * NN + i] = (tm - mtm) * rtm;
  }

  // alpha/beta per kernel dim, pre-negated and pre-scaled by log2(e) for exp2
  if (tid < DD) {
    float s0 = __expf(kls[tid * 2 + 0]);
    float s1 = __expf(kls[tid * 2 + 1]);
    float tr = s0 + s1;
    float bw = erfinv_f((float)(tid + 1) / (float)(DD + 1)) * SQRT2F;
    float c  = -LOG2EF / (2.0f * bw * bw * tr);
    ws[2 * BB * NN + tid]      = s0 * c;
    ws[2 * BB * NN + DD + tid] = s1 * c;
  }

  // ext = bn of 5 market series over B=4
  if (tid < 5) {
    const float* p = (tid == 0) ? mkt_ret : (tid == 1) ? mkt_vol
                   : (tid == 2) ? tsy     : (tid == 3) ? tvm : tvs;
    float m = 0.f;
    for (int b = 0; b < BB; ++b) m += p[b];
    m *= 0.25f;
    float v = 0.f;
    for (int b = 0; b < BB; ++b) { float d = p[b] - m; v += d * d; }
    v *= 0.25f;
    float r = rsqrtf(v + EPSF);
    for (int b = 0; b < BB; ++b) ext_out[b * 5 + tid] = (p[b] - m) * r;
  }
}

// ---------------- Stage 2: WMMA Nadaraya-Watson + LN + PE + final LN ---------
__global__ __launch_bounds__(256)
void se_main(const float* __restrict__ ws,
             const float* __restrict__ surf_tv,
             const float* __restrict__ pe_ls,
             const float* __restrict__ mask_token,
             const float* __restrict__ ln_in_g, const float* __restrict__ ln_in_b,
             const float* __restrict__ ln_q_g,  const float* __restrict__ ln_q_b,
             const float* __restrict__ ln_f_g,  const float* __restrict__ ln_f_b,
             float* __restrict__ out) {
  __shared__ float    s_lm[SS], s_tm[SS];
  __shared__ _Float16 s_tvh[SS];
  __shared__ float    s_alpha[DD], s_beta[DD];
  __shared__ float    s_par[7][DD];        // in_g,in_b,q_g,q_b,f_g,f_b,mask
  __shared__ float    s_wk[8][4][2][16];   // wave, d-block, {weighted,ksum}, m

  const int tid  = threadIdx.x;
  const int b    = blockIdx.x / (NN / 8);
  const int tile = blockIdx.x % (NN / 8);

  const float* nlm_all = ws;
  const float* ntm_all = ws + BB * NN;
  const float* alpha_g = ws + 2 * BB * NN;
  const float* beta_g  = alpha_g + DD;

  for (int s = tid; s < SS; s += 256) {
    s_lm[s]  = nlm_all[b * NN + s];
    s_tm[s]  = ntm_all[b * NN + s];
    s_tvh[s] = (_Float16)surf_tv[b * SS + s];
  }
  if (tid < DD) {
    s_alpha[tid]  = alpha_g[tid];
    s_beta[tid]   = beta_g[tid];
    s_par[0][tid] = ln_in_g[tid];  s_par[1][tid] = ln_in_b[tid];
    s_par[2][tid] = ln_q_g[tid];   s_par[3][tid] = ln_q_b[tid];
    s_par[4][tid] = ln_f_g[tid];   s_par[5][tid] = ln_f_b[tid];
    s_par[6][tid] = mask_token[tid];
  }
  __syncthreads();

  const int wave = tid >> 5;
  const int lane = tid & 31;
  const int col  = lane & 15;               // A-row within d-block; also B column
  const int kbA  = (lane < 16) ? 0 : 8;     // 16-bit A 16x32 lane->K mapping
  const int kbB  = (lane < 16) ? 0 : 8;     // 16-bit B 32x16 lane->K mapping (same as A)
  const int n    = tile * 8 + wave;

  const float nlm = nlm_all[b * NN + n];
  const float ntm = ntm_all[b * NN + n];
  const float ls  = pe_ls[0];

  // branchless B-column selectors (lane constants)
  const _Float16 hsel = (col == 0) ? (_Float16)1.0f : (_Float16)0.0f;
  const _Float16 hone = (col == 1) ? (_Float16)1.0f : (_Float16)0.0f;

  // per-lane alpha/beta for the 4 d-blocks (already -log2e scaled)
  float al[4], be[4];
#pragma unroll
  for (int db = 0; db < 4; ++db) {
    al[db] = s_alpha[db * 16 + col];
    be[db] = s_beta[db * 16 + col];
  }

  // --- Gaussian kernel rows -> f16, reduced over S by v_wmma_f32_16x16x32_f16
  v8f acc[4];
#pragma unroll
  for (int db = 0; db < 4; ++db) acc[db] = (v8f){};

  for (int s0 = 0; s0 < SS; s0 += 32) {
    // shared-across-d work: squared coordinate diffs for this 32-wide s slice
    float du2[16], dv2[16];
#pragma unroll
    for (int j = 0; j < 16; ++j) {
      // lanes 0-15: K in {0..7,16..23}; lanes 16-31: K in {8..15,24..31}
      int s = s0 + kbA + j + ((j < 8) ? 0 : 8);
      float du = nlm - s_lm[s];
      float dv = ntm - s_tm[s];
      du2[j] = du * du;
      dv2[j] = dv * dv;
    }
    // B = [tv | 1 | 0 ... 0], built branchlessly once per slice
    v16h bm;
#pragma unroll
    for (int t = 0; t < 16; ++t)
      bm[t] = s_tvh[s0 + kbB + t + ((t < 8) ? 0 : 8)] * hsel + hone;

#pragma unroll
    for (int db = 0; db < 4; ++db) {
      v16h a;
#pragma unroll
      for (int j = 0; j < 16; ++j)
        a[j] = (_Float16)exp2f(fmaf(al[db], du2[j], be[db] * dv2[j]));
      acc[db] = __builtin_amdgcn_wmma_f32_16x16x32_f16(false, a, false, bm,
                                                       (short)0, acc[db], false, false);
    }
  }

  // C/D layout: lane L, VGPR r holds (M = r + (L<16?0:8), N = L&15)
  if (col < 2) {
    int mhi = (lane < 16) ? 0 : 8;
#pragma unroll
    for (int db = 0; db < 4; ++db)
#pragma unroll
      for (int r = 0; r < 8; ++r)
        s_wk[wave][db][col][mhi + r] = acc[db][r];
  }
  __syncthreads();

  // --- per-row LN -> + sqrt2 * PE (+ mask for queries) -> final LN ----------
  const int d0 = lane, d1 = lane + 32;
  const int m  = lane & 15;
  const int dbl = lane >> 4;                 // d0 lives in block 0/1, d1 in 2/3
  float e0 = s_wk[wave][dbl][0][m]     / s_wk[wave][dbl][1][m];
  float e1 = s_wk[wave][2 + dbl][0][m] / s_wk[wave][2 + dbl][1][m];

  float sum = e0 + e1, sq = e0 * e0 + e1 * e1;
#pragma unroll
  for (int off = 16; off > 0; off >>= 1) {
    sum += __shfl_xor(sum, off, 32);
    sq  += __shfl_xor(sq,  off, 32);
  }
  float mean = sum * (1.0f / DD);
  float rstd = rsqrtf(sq * (1.0f / DD) - mean * mean + EPSF);

  const bool isq = (n >= SS);
  float g0 = isq ? s_par[2][d0] : s_par[0][d0];
  float b0 = isq ? s_par[3][d0] : s_par[1][d0];
  float g1 = isq ? s_par[2][d1] : s_par[0][d1];
  float b1 = isq ? s_par[3][d1] : s_par[1][d1];

  // interleaved sin/cos positional embedding: pe[4i+{0,1,2,3}] = sin(x),cos(x),sin(t),cos(t)
  auto pe_val = [&](int d) -> float {
    int i = d >> 2, cc = d & 3;
    float rdiv = __expf(-ls * (float)i * 0.0625f);       // 1 / exp(ls)^(4i/D)
    float arg  = ((cc < 2) ? nlm : ntm) * rdiv;
    return (cc & 1) ? __cosf(arg) : __sinf(arg);
  };

  float t0 = (e0 - mean) * rstd * g0 + b0 + pe_val(d0) * SQRT2F + (isq ? s_par[6][d0] : 0.0f);
  float t1 = (e1 - mean) * rstd * g1 + b1 + pe_val(d1) * SQRT2F + (isq ? s_par[6][d1] : 0.0f);

  sum = t0 + t1; sq = t0 * t0 + t1 * t1;
#pragma unroll
  for (int off = 16; off > 0; off >>= 1) {
    sum += __shfl_xor(sum, off, 32);
    sq  += __shfl_xor(sq,  off, 32);
  }
  float mean2 = sum * (1.0f / DD);
  float rstd2 = rsqrtf(sq * (1.0f / DD) - mean2 * mean2 + EPSF);

  float* row = out + (size_t)b * NN * DD + (size_t)n * DD;
  row[d0] = (t0 - mean2) * rstd2 * s_par[4][d0] + s_par[5][d0];
  row[d1] = (t1 - mean2) * rstd2 * s_par[4][d1] + s_par[5][d1];
}

extern "C" void kernel_launch(void* const* d_in, const int* in_sizes, int n_in,
                              void* d_out, int out_size, void* d_ws, size_t ws_size,
                              hipStream_t stream) {
  const float* surf_lm  = (const float*)d_in[0];
  const float* surf_ttm = (const float*)d_in[1];
  const float* surf_tv  = (const float*)d_in[2];
  const float* q_lm     = (const float*)d_in[3];
  const float* q_ttm    = (const float*)d_in[4];
  const float* mkt_ret  = (const float*)d_in[5];
  const float* mkt_vol  = (const float*)d_in[6];
  const float* tsy_rate = (const float*)d_in[7];
  const float* tv_mean  = (const float*)d_in[8];
  const float* tv_std   = (const float*)d_in[9];
  const float* kls      = (const float*)d_in[10];
  const float* pe_ls    = (const float*)d_in[11];
  const float* mask_tok = (const float*)d_in[12];
  const float* ln_in_g  = (const float*)d_in[13];
  const float* ln_in_b  = (const float*)d_in[14];
  const float* ln_q_g   = (const float*)d_in[15];
  const float* ln_q_b   = (const float*)d_in[16];
  const float* ln_f_g   = (const float*)d_in[17];
  const float* ln_f_b   = (const float*)d_in[18];

  float* out = (float*)d_out;
  float* ws  = (float*)d_ws;  // needs 6272 floats = 25 KB

  se_stage1<<<1, 256, 0, stream>>>(surf_lm, surf_ttm, q_lm, q_ttm,
                                   mkt_ret, mkt_vol, tsy_rate, tv_mean, tv_std,
                                   kls, ws, out + BB * NN * DD);

  se_main<<<BB * (NN / 8), 256, 0, stream>>>(ws, surf_tv, pe_ls, mask_tok,
                                             ln_in_g, ln_in_b, ln_q_g, ln_q_b,
                                             ln_f_g, ln_f_b, out);
}
